// CMAttention_34522947125894
// MI455X (gfx1250) — matrix operand order repaired and checked by reading, not compiled, not run on hardware
//
#include <hip/hip_runtime.h>

typedef _Float16 v16h __attribute__((ext_vector_type(16)));
typedef _Float16 v8h  __attribute__((ext_vector_type(8)));
typedef _Float16 v4h  __attribute__((ext_vector_type(4)));
typedef float    v8f  __attribute__((ext_vector_type(8)));

// ---------------------------------------------------------------------------
// WMMA helpers (CDNA5 gfx1250, wave32): D = A(16x32 f16) x B(32x16 f16) + C(f32)
// ---------------------------------------------------------------------------
__device__ __forceinline__ v8f wmma_f16(v16h a, v16h b, v8f c) {
    return __builtin_amdgcn_wmma_f32_16x16x32_f16(
        /*neg_a=*/false, a, /*neg_b=*/false, b,
        /*c_mod=*/(short)0, c, /*reuse_a=*/false, /*reuse_b=*/false);
}

// A-matrix 16x32 f16 layout (ISA 7.12.2): lane-half hh holds K in
// [8*hh .. 8*hh+7] and [16+8*hh .. 23+8*hh]; row M = lane&15, K-contiguous src.
__device__ __forceinline__ v16h load_afrag(const _Float16* rowk, int hh) {
    union { v16h v; v8h h2[2]; } u;
    u.h2[0] = *(const v8h*)(rowk + 8 * hh);
    u.h2[1] = *(const v8h*)(rowk + 16 + 8 * hh);
    return u.v;
}
// B-matrix 32x16 f16: lane-half hh holds K in [8*hh .. 8*hh+7] and
// [16+8*hh .. 23+8*hh], column N = lane&15; src points at K-contiguous column data.
__device__ __forceinline__ v16h load_bfrag(const _Float16* colk, int hh) {
    union { v16h v; v8h h2[2]; } u;
    u.h2[0] = *(const v8h*)(colk + 8 * hh);
    u.h2[1] = *(const v8h*)(colk + 16 + 8 * hh);
    return u.v;
}
__device__ __forceinline__ v8f zero8() {
    v8f z = {0.f, 0.f, 0.f, 0.f, 0.f, 0.f, 0.f, 0.f};
    return z;
}

// ---------------------------------------------------------------------------
// CDNA5 async memory->LDS copy (ASYNCcnt path, ISA 10.x / 15.18).
// Generic pointers into the LDS aperture carry the LDS byte offset in their
// low 32 bits.
// ---------------------------------------------------------------------------
__device__ __forceinline__ void async_copy_b128(void* lds_dst, const void* gsrc) {
    unsigned int l = (unsigned int)(unsigned long long)lds_dst;
    asm volatile("global_load_async_to_lds_b128 %0, %1, off"
                 :: "v"(l), "v"(gsrc) : "memory");
}
__device__ __forceinline__ void wait_async0() {
    asm volatile("s_wait_asynccnt 0" ::: "memory");
}

// ---------------------------------------------------------------------------
// Kernel 1: qkv = X[4096,512] @ W[512,1536] for (x,Wqkv_x) and (a,Wqkv_a)
// Block tile 128x128, 8 waves, each wave 32(M)x64(N); K-step 32.
// Staging: float4 loads, packed f16 b64 LDS stores (W via 4x4 reg transpose).
// ---------------------------------------------------------------------------
__global__ __launch_bounds__(256) void qkv_gemm(
    const float* __restrict__ X0, const float* __restrict__ W0, float* __restrict__ C0,
    const float* __restrict__ X1, const float* __restrict__ W1, float* __restrict__ C1)
{
    const float* X = blockIdx.z ? X1 : X0;
    const float* W = blockIdx.z ? W1 : W0;
    float*       C = blockIdx.z ? C1 : C0;

    __shared__ __align__(64) _Float16 Xs[128][32];   // [m][k]
    __shared__ __align__(64) _Float16 Ws[128][32];   // [n][k] (transposed)

    const int tid  = threadIdx.x;
    const int lane = tid & 31, wave = tid >> 5;
    const int wm = wave >> 1, wn = wave & 1;        // 4x2 wave grid
    const int hh = lane >> 4, ln = lane & 15;
    const int m0 = blockIdx.y * 128, n0 = blockIdx.x * 128;

    v8f acc[2][4];
    for (int ta = 0; ta < 2; ++ta)
        for (int tb = 0; tb < 4; ++tb) acc[ta][tb] = zero8();

    const int xw_r  = tid >> 3;            // X: row, 8 float4 per row
    const int xw_c  = (tid & 7) * 4;       // X: col base
    const int wk    = (tid >> 5) * 4;      // W: 4-row k block
    const int wn4   = (tid & 31) * 4;      // W: 4-col n block

    for (int k0 = 0; k0 < 512; k0 += 32) {
        __syncthreads();
        // X tile: 1024 float4; each thread 4 (rows tid>>3, +32, +64, +96)
        #pragma unroll
        for (int s = 0; s < 4; ++s) {
            int r = xw_r + 32 * s;
            float4 f = *(const float4*)&X[(size_t)(m0 + r) * 512 + k0 + xw_c];
            v4h h4 = {(_Float16)f.x, (_Float16)f.y, (_Float16)f.z, (_Float16)f.w};
            *(v4h*)&Xs[r][xw_c] = h4;
        }
        // W tile: each thread transposes a 4(k) x 4(n) block
        {
            const float* wp = &W[(size_t)(k0 + wk) * 1536 + n0 + wn4];
            float4 f0 = *(const float4*)(wp);
            float4 f1 = *(const float4*)(wp + 1536);
            float4 f2 = *(const float4*)(wp + 3072);
            float4 f3 = *(const float4*)(wp + 4608);
            v4h c0 = {(_Float16)f0.x, (_Float16)f1.x, (_Float16)f2.x, (_Float16)f3.x};
            v4h c1 = {(_Float16)f0.y, (_Float16)f1.y, (_Float16)f2.y, (_Float16)f3.y};
            v4h c2 = {(_Float16)f0.z, (_Float16)f1.z, (_Float16)f2.z, (_Float16)f3.z};
            v4h c3 = {(_Float16)f0.w, (_Float16)f1.w, (_Float16)f2.w, (_Float16)f3.w};
            *(v4h*)&Ws[wn4 + 0][wk] = c0;
            *(v4h*)&Ws[wn4 + 1][wk] = c1;
            *(v4h*)&Ws[wn4 + 2][wk] = c2;
            *(v4h*)&Ws[wn4 + 3][wk] = c3;
        }
        __syncthreads();

        v16h afr[2], bfr[4];
        #pragma unroll
        for (int ta = 0; ta < 2; ++ta)
            afr[ta] = load_afrag(&Xs[32 * wm + 16 * ta + ln][0], hh);
        #pragma unroll
        for (int tb = 0; tb < 4; ++tb)
            bfr[tb] = load_bfrag(&Ws[64 * wn + 16 * tb + ln][0], hh);
        #pragma unroll
        for (int ta = 0; ta < 2; ++ta)
            #pragma unroll
            for (int tb = 0; tb < 4; ++tb)
                acc[ta][tb] = wmma_f16(afr[ta], bfr[tb], acc[ta][tb]);
    }

    for (int ta = 0; ta < 2; ++ta)
        for (int tb = 0; tb < 4; ++tb)
            for (int r = 0; r < 8; ++r) {
                int row = m0 + 32 * wm + 16 * ta + r + 8 * hh;
                int col = n0 + 64 * wn + 16 * tb + ln;
                C[(size_t)row * 1536 + col] = acc[ta][tb][r];
            }
}

// ---------------------------------------------------------------------------
// Kernel 2: LayerNorm(q_x,k_x,q_a,k_a) + concat + RoPE + pack to heads (f16).
// One block per (pos, b). Q gets 1/sqrt(128) folded in. V stored transposed.
// ---------------------------------------------------------------------------
__global__ __launch_bounds__(256) void ln_rope_pack(
    const float* __restrict__ qkvx, const float* __restrict__ qkva,
    const float* __restrict__ g_qx, const float* __restrict__ b_qx,
    const float* __restrict__ g_kx, const float* __restrict__ b_kx,
    const float* __restrict__ g_qa, const float* __restrict__ b_qa,
    const float* __restrict__ g_ka, const float* __restrict__ b_ka,
    _Float16* __restrict__ Qh, _Float16* __restrict__ Kh, _Float16* __restrict__ Vt)
{
    const int pos = blockIdx.x, b = blockIdx.y, t = threadIdx.x;
    __shared__ float qln[1024], kln[1024];
    __shared__ float sS[256], sS2[256];

    const float* bx = qkvx + ((size_t)b * 1024 + pos) * 1536;
    const float* ba = qkva + ((size_t)b * 1024 + pos) * 1536;

    const float* srcs[4] = { bx, bx + 512, ba, ba + 512 };
    const float* gams[4] = { g_qx, g_kx, g_qa, g_ka };
    const float* bets[4] = { b_qx, b_kx, b_qa, b_ka };
    float*       dsts[4] = { qln, kln, qln + 512, kln + 512 };

    for (int g = 0; g < 4; ++g) {
        float x0 = srcs[g][t], x1 = srcs[g][t + 256];
        sS[t] = x0 + x1;
        sS2[t] = x0 * x0 + x1 * x1;
        __syncthreads();
        for (int off = 128; off > 0; off >>= 1) {
            if (t < off) { sS[t] += sS[t + off]; sS2[t] += sS2[t + off]; }
            __syncthreads();
        }
        float mu  = sS[0] * (1.f / 512.f);
        float var = sS2[0] * (1.f / 512.f) - mu * mu;
        float rs  = rsqrtf(var + 1e-5f);
        dsts[g][t]       = (x0 - mu) * rs * gams[g][t]       + bets[g][t];
        dsts[g][t + 256] = (x1 - mu) * rs * gams[g][t + 256] + bets[g][t + 256];
        __syncthreads();
    }

    const float qscale = 0.08838834764831845f;          // 1/sqrt(128)
    const float lnf    = 0.14391156840f;                // ln(10000)/64
    for (int idx = t; idx < 1024; idx += 256) {
        int hd = idx >> 7, d = idx & 127, i = d & 63;
        float ang = (float)pos * __expf(-(float)i * lnf);
        float sn, cs;
        sincosf(ang, &sn, &cs);
        int pidx = hd * 128 + ((d < 64) ? d + 64 : d - 64);
        float qv = qln[idx], kv = kln[idx];
        float qp = qln[pidx], kp = kln[pidx];
        if (d < 64) { qp = -qp; kp = -kp; }
        float qr = qv * cs + qp * sn;
        float kr = kv * cs + kp * sn;
        size_t bh = (size_t)(b * 8 + hd);
        Qh[(bh * 1024 + pos) * 128 + d] = (_Float16)(qr * qscale);
        Kh[(bh * 1024 + pos) * 128 + d] = (_Float16)kr;
        float vv = (idx < 512) ? bx[1024 + idx] : ba[1024 + idx - 512];
        Vt[(bh * 128 + d) * 1024 + pos] = (_Float16)vv;   // V transposed [d][pos]
    }
}

// ---------------------------------------------------------------------------
// Kernel 3: flash attention. Block = (b,h) x 128 q-rows; 8 waves x 16 rows.
// K/V tiles of 32 positions double-buffered in LDS via CDNA5 async copies;
// S and O both via WMMA.
// ---------------------------------------------------------------------------
__global__ __launch_bounds__(256) void attn(
    const _Float16* __restrict__ Qh, const _Float16* __restrict__ Kh,
    const _Float16* __restrict__ Vt, float* __restrict__ Out)
{
    __shared__ __align__(64) _Float16 Ks[2][32][128];   // [buf][kpos][d]
    __shared__ __align__(64) _Float16 Vs[2][128][32];   // [buf][d][kpos]
    __shared__ __align__(64) _Float16 Ps[8][16][32];    // per-wave P staging

    const int tid = threadIdx.x, lane = tid & 31, wave = tid >> 5;
    const int hh = lane >> 4, ln = lane & 15;
    const int bh = blockIdx.y;
    const int b = bh >> 3, hd = bh & 7;
    const int qbase = blockIdx.x * 128 + wave * 16;

    const _Float16* Qg = Qh + ((size_t)bh * 1024 + qbase) * 128;
    const _Float16* Kg = Kh + (size_t)bh * 1024 * 128;
    const _Float16* Vg = Vt + (size_t)bh * 128 * 1024;

    v16h qfr[4];
    #pragma unroll
    for (int c = 0; c < 4; ++c)
        qfr[c] = load_afrag(Qg + (size_t)ln * 128 + 32 * c, hh);

    v8f o[8];
    float m_i[8], l_i[8];
    for (int j = 0; j < 8; ++j) o[j] = zero8();
    for (int r = 0; r < 8; ++r) { m_i[r] = -1e30f; l_i[r] = 0.f; }

    _Float16* Pw = &Ps[wave][0][0];

    // per-thread V-tile copy coordinates (2 x b128 per thread)
    const int vd0 = tid >> 2,          vg0 = (tid & 3) * 8;
    const int vd1 = (tid + 256) >> 2,  vg1 = ((tid + 256) & 3) * 8;

    auto issue_tile = [&](int kt, int bufi) {
        const _Float16* srcK = Kg + (size_t)kt * 32 * 128;   // 4096 halves, contiguous
        _Float16* dstK = &Ks[bufi][0][0];
        async_copy_b128(dstK + (size_t)tid * 8,         srcK + (size_t)tid * 8);
        async_copy_b128(dstK + (size_t)(tid + 256) * 8, srcK + (size_t)(tid + 256) * 8);
        async_copy_b128(&Vs[bufi][vd0][vg0], Vg + (size_t)vd0 * 1024 + kt * 32 + vg0);
        async_copy_b128(&Vs[bufi][vd1][vg1], Vg + (size_t)vd1 * 1024 + kt * 32 + vg1);
    };

    issue_tile(0, 0);

    for (int kt = 0; kt < 32; ++kt) {
        const int cur = kt & 1;
        wait_async0();          // this wave's async writes (tile kt) landed
        __syncthreads();        // everyone's landed; prev reads of other buf done
        if (kt + 1 < 32) issue_tile(kt + 1, cur ^ 1);

        // S = Q(16x128) @ K^T(128x32): two 16-col tiles, K-dim chunks of 32
        v8f s0 = zero8(), s1 = zero8();
        #pragma unroll
        for (int c = 0; c < 4; ++c) {
            v16h b0 = load_bfrag(&Ks[cur][ln][32 * c], hh);
            v16h b1 = load_bfrag(&Ks[cur][16 + ln][32 * c], hh);
            s0 = wmma_f16(qfr[c], b0, s0);
            s1 = wmma_f16(qfr[c], b1, s1);
        }

        // online softmax (row r+8*hh lives in VGPR r; cols spread over 16 lanes)
        float p0[8], p1[8], alpha[8];
        #pragma unroll
        for (int r = 0; r < 8; ++r) {
            float v = fmaxf(s0[r], s1[r]);
            v = fmaxf(v, __shfl_xor(v, 1));
            v = fmaxf(v, __shfl_xor(v, 2));
            v = fmaxf(v, __shfl_xor(v, 4));
            v = fmaxf(v, __shfl_xor(v, 8));
            float mn = fmaxf(m_i[r], v);
            alpha[r] = __expf(m_i[r] - mn);
            m_i[r] = mn;
            p0[r] = __expf(s0[r] - mn);
            p1[r] = __expf(s1[r] - mn);
            float rs = p0[r] + p1[r];
            rs += __shfl_xor(rs, 1);
            rs += __shfl_xor(rs, 2);
            rs += __shfl_xor(rs, 4);
            rs += __shfl_xor(rs, 8);
            l_i[r] = l_i[r] * alpha[r] + rs;
        }
        #pragma unroll
        for (int j = 0; j < 8; ++j)
            #pragma unroll
            for (int r = 0; r < 8; ++r) o[j][r] *= alpha[r];

        // C-layout -> A-fragment layout via per-wave LDS staging (f16 round)
        #pragma unroll
        for (int r = 0; r < 8; ++r) {
            Pw[(r + 8 * hh) * 32 + ln]      = (_Float16)p0[r];
            Pw[(r + 8 * hh) * 32 + 16 + ln] = (_Float16)p1[r];
        }
        v16h pa = load_afrag(Pw + ln * 32, hh);

        // O += P(16x32) @ V(32x128): 8 d-tiles of 16
        #pragma unroll
        for (int j = 0; j < 8; ++j) {
            v16h bv = load_bfrag(&Vs[cur][16 * j + ln][0], hh);
            o[j] = wmma_f16(pa, bv, o[j]);
        }
    }

    float inv[8];
    for (int r = 0; r < 8; ++r) inv[r] = 1.f / l_i[r];
    for (int j = 0; j < 8; ++j)
        for (int r = 0; r < 8; ++r) {
            int row = qbase + r + 8 * hh;
            int col = hd * 128 + 16 * j + ln;
            Out[((size_t)b * 1024 + row) * 1024 + col] = o[j][r] * inv[r];
        }
}

// ---------------------------------------------------------------------------
extern "C" void kernel_launch(void* const* d_in, const int* in_sizes, int n_in,
                              void* d_out, int out_size, void* d_ws, size_t ws_size,
                              hipStream_t stream)
{
    const float* x    = (const float*)d_in[0];
    const float* a    = (const float*)d_in[1];
    const float* Wx   = (const float*)d_in[2];
    const float* Wa   = (const float*)d_in[3];
    const float* g_qx = (const float*)d_in[4];
    const float* b_qx = (const float*)d_in[5];
    const float* g_kx = (const float*)d_in[6];
    const float* b_kx = (const float*)d_in[7];
    const float* g_qa = (const float*)d_in[8];
    const float* b_qa = (const float*)d_in[9];
    const float* g_ka = (const float*)d_in[10];
    const float* b_ka = (const float*)d_in[11];
    float* out = (float*)d_out;

    // Workspace layout (~72 MB):
    //   qkv_x f32 [4,1024,1536], qkv_a f32 [4,1024,1536],
    //   Q f16 [4,8,1024,128], K f16 [4,8,1024,128], V^T f16 [4,8,128,1024]
    const size_t QKV_ELEMS = (size_t)4 * 1024 * 1536;
    const size_t HEAD_ELEMS = (size_t)4 * 8 * 1024 * 128;
    float* qkvx = (float*)d_ws;
    float* qkva = qkvx + QKV_ELEMS;
    _Float16* Qh = (_Float16*)(qkva + QKV_ELEMS);
    _Float16* Kh = Qh + HEAD_ELEMS;
    _Float16* Vt = Kh + HEAD_ELEMS;

    qkv_gemm<<<dim3(12, 32, 2), 256, 0, stream>>>(x, Wx, qkvx, a, Wa, qkva);
    ln_rope_pack<<<dim3(1024, 4), 256, 0, stream>>>(
        qkvx, qkva, g_qx, b_qx, g_kx, b_kx, g_qa, b_qa, g_ka, b_ka, Qh, Kh, Vt);
    attn<<<dim3(8, 32), 256, 0, stream>>>(Qh, Kh, Vt, out);
}
